// Reservoir_45999099740380
// MI455X (gfx1250) — matrix operand order repaired and checked by reading, not compiled, not run on hardware
//
#include <hip/hip_runtime.h>

// ---------------- problem constants (match reference) ----------------
#define NN   4096
#define TT   2048
#define DT_C 0.001f
#define TAU_C 10.0f
#define VTH_C 100.0f
#define VR_C  (-100.0f)

typedef __attribute__((ext_vector_type(16))) _Float16 v16h;
typedef __attribute__((ext_vector_type(8)))  float    v8f;

// ---------------------------------------------------------------------
// Kernel 1: pack W (f32, row-major NxN) into f16 tiles laid out exactly
// as the per-lane A-fragment of V_WMMA_F32_16X16X32_F16.
// Tile = 16 rows x 32 cols, contiguous K order (matches B fragment):
//   lanes 0-15 : row = lane,      K = {0..15}
//   lanes 16-31: row = lane-16,   K = {16..31}
//   slot order within a lane: K0..K15 (resp. +16)
// Storage: tile-major [rowblk][colblk]; inside a tile lane-major
// (32 lanes x 16 halves = 1 KB), so the A load is lane*32B contiguous.
// ---------------------------------------------------------------------
__global__ __launch_bounds__(256) void pack_w_kernel(
    const float* __restrict__ W, _Float16* __restrict__ Wp) {
  int idx = blockIdx.x * 256 + threadIdx.x;    // one element of W
  int i = idx >> 12;                           // row   (N = 4096)
  int j = idx & (NN - 1);                      // col
  int r  = i >> 4,  m  = i & 15;               // row-block, row-in-block
  int c  = j >> 5,  kk = j & 31;               // col-block, k-in-block
  int grp  = kk >> 4;                          // which lane half holds it
  int slot = kk & 15;                          // half index inside lane
  int lane = m + (grp << 4);
  size_t tile = (size_t)r * 128 + c;           // 128 k-tiles per row-block
  Wp[tile * 512 + lane * 16 + slot] = (_Float16)W[idx];
}

// ---------------------------------------------------------------------
// Kernel 2: one reservoir timestep.
//   256 blocks x 256 threads (8 waves of 32, wave32).
//   Block b owns rows [b*16, b*16+16): one 16-row block.
//   wave w handles K-chunk w: 16 tiles of K=32 (total 128 tiles = K 4096).
// B-fragment: spikes replicated over the 16 N columns. In this code's
// contiguous K striping (lanes 0-15 -> K=0..15, lanes 16-31 -> K=16..31),
// each lane's v16h is one contiguous 32B run of the spike array in LDS.
// Four independent accumulators keep >=4 tile loads in flight and break
// the WMMA RAW chain; partial dots reduced across the 8 waves via LDS.
// ---------------------------------------------------------------------
__global__ __launch_bounds__(256) void reservoir_step_kernel(
    const float*    __restrict__ x_t,     // x[t, :]   (N floats)
    const _Float16* __restrict__ Wp,      // packed W  (N*N halves)
    const float*    __restrict__ u_in,
    const float*    __restrict__ s_in,
    float*          __restrict__ u_out,
    float*          __restrict__ s_out,
    float*          __restrict__ trace)   // s_trace[t, :]
{
  __shared__ __align__(32) _Float16 sp[NN];   // spike mask, 8 KB
  __shared__ float part[8][16];               // per-wave partial dots

  const int tid  = threadIdx.x;
  const int wave = tid >> 5;
  const int lane = tid & 31;

  // ---- phase 1: spikes = (u_old >= v_th) as f16 {0,1}, full vector ----
  {
    const int j0 = tid * 16;
    const float4* up = (const float4*)(u_in + j0);
    v16h m;
#pragma unroll
    for (int q = 0; q < 4; ++q) {
      float4 v = up[q];
      m[q * 4 + 0] = (_Float16)((v.x >= VTH_C) ? 1.0f : 0.0f);
      m[q * 4 + 1] = (_Float16)((v.y >= VTH_C) ? 1.0f : 0.0f);
      m[q * 4 + 2] = (_Float16)((v.z >= VTH_C) ? 1.0f : 0.0f);
      m[q * 4 + 3] = (_Float16)((v.w >= VTH_C) ? 1.0f : 0.0f);
    }
    *(v16h*)(&sp[j0]) = m;                    // 2x ds_store_b128
  }
  __syncthreads();

  // ---- phase 2: WMMA matvec  dot = W[rows,:] @ spikes ----
  const int rb = blockIdx.x;                         // row-block 0..255
  const _Float16* abase =
      Wp + ((size_t)rb * 128 + (size_t)wave * 16) * 512 + lane * 16;

  v8f acc0 = {}, acc1 = {}, acc2 = {}, acc3 = {};
  const int bhalf = (lane & 16);                     // 0 or 16 (K offset)
  const int jbase = ((wave * 16) << 5) + bhalf;      // spike index base
  for (int c = 0; c < 16; c += 4) {
    v16h a0 = *(const v16h*)(abase + (size_t)(c + 0) * 512);
    v16h b0 = *(const v16h*)(&sp[jbase + ((c + 0) << 5)]);
    acc0 = __builtin_amdgcn_wmma_f32_16x16x32_f16(
        false, a0, false, b0, (short)0, acc0, false, false);

    v16h a1 = *(const v16h*)(abase + (size_t)(c + 1) * 512);
    v16h b1 = *(const v16h*)(&sp[jbase + ((c + 1) << 5)]);
    acc1 = __builtin_amdgcn_wmma_f32_16x16x32_f16(
        false, a1, false, b1, (short)0, acc1, false, false);

    v16h a2 = *(const v16h*)(abase + (size_t)(c + 2) * 512);
    v16h b2 = *(const v16h*)(&sp[jbase + ((c + 2) << 5)]);
    acc2 = __builtin_amdgcn_wmma_f32_16x16x32_f16(
        false, a2, false, b2, (short)0, acc2, false, false);

    v16h a3 = *(const v16h*)(abase + (size_t)(c + 3) * 512);
    v16h b3 = *(const v16h*)(&sp[jbase + ((c + 3) << 5)]);
    acc3 = __builtin_amdgcn_wmma_f32_16x16x32_f16(
        false, a3, false, b3, (short)0, acc3, false, false);

    // WGP-scope prefetch (default scope) of upcoming W tiles into the
    // near cache; builtin __builtin_prefetch only reaches SE/SYS scope.
    asm volatile("global_prefetch_b8 %0, off"
                 :: "v"((const void*)(abase + (size_t)(c + 4) * 512)));
  }
  v8f acc = (acc0 + acc1) + (acc2 + acc3);

  // D layout: lane 0 holds rows 0..7 (VGPR q -> M=q), lane 16 rows 8..15.
  if (lane == 0) {
#pragma unroll
    for (int q = 0; q < 8; ++q) part[wave][q] = acc[q];
  } else if (lane == 16) {
#pragma unroll
    for (int q = 0; q < 8; ++q) part[wave][8 + q] = acc[q];
  }
  __syncthreads();

  // ---- phase 3: reduce K-chunks + elementwise update (reference order) ----
  if (tid < 16) {
    float d = ((part[0][tid] + part[1][tid]) + (part[2][tid] + part[3][tid])) +
              ((part[4][tid] + part[5][tid]) + (part[6][tid] + part[7][tid]));
    int i = (blockIdx.x << 4) + tid;           // global neuron index

    float uo = u_in[i];
    float so = s_in[i];
    float spike = (uo >= VTH_C) ? 1.0f : 0.0f;            // OLD u
    float un = uo + DT_C * (uo * uo + x_t[i] + so);       // uses OLD s
    float sn = so + DT_C * (-so / TAU_C) + d;             // + W @ spikes
    un = (spike > 0.0f) ? VR_C : un;                      // reset, OLD mask

    s_out[i]  = sn;
    u_out[i]  = un;
    trace[i]  = sn;
  }
}

// ---------------------------------------------------------------------
// Host launcher: pack once, then 2048 dependent step kernels on `stream`.
// Scratch layout: [0, 32MB) packed f16 W; then u double-buffer (2N f32),
// then s double-buffer (2N f32). All scratch is fully (re)written before
// being read each call -> deterministic, poison-safe, capture-safe.
// ---------------------------------------------------------------------
extern "C" void kernel_launch(void* const* d_in, const int* in_sizes, int n_in,
                              void* d_out, int out_size, void* d_ws, size_t ws_size,
                              hipStream_t stream) {
  (void)in_sizes; (void)n_in; (void)out_size; (void)ws_size;

  const float* x  = (const float*)d_in[0];   // [T, N]
  const float* W  = (const float*)d_in[1];   // [N, N]
  const float* u0 = (const float*)d_in[2];   // [N]
  const float* s0 = (const float*)d_in[3];   // [N]
  float* out = (float*)d_out;                // [T, N] s_trace

  char* ws = (char*)d_ws;
  _Float16* Wp  = (_Float16*)ws;                                   // 32 MB
  float* ubuf = (float*)(ws + (size_t)NN * NN * sizeof(_Float16)); // 2*N
  float* sbuf = ubuf + 2 * NN;                                     // 2*N

  // one-time pack: W f32 -> f16 WMMA-A-fragment tiles
  pack_w_kernel<<<(NN * NN) / 256, 256, 0, stream>>>(W, Wp);

  for (int t = 0; t < TT; ++t) {
    const float* u_in = (t == 0) ? u0 : ubuf + (((t & 1) ^ 1) * NN);
    const float* s_in = (t == 0) ? s0 : sbuf + (((t & 1) ^ 1) * NN);
    float* u_out = ubuf + ((t & 1) * NN);
    float* s_out = sbuf + ((t & 1) * NN);
    reservoir_step_kernel<<<256, 256, 0, stream>>>(
        x + (size_t)t * NN, Wp, u_in, s_in, u_out, s_out,
        out + (size_t)t * NN);
  }
}
